// ConditionalSmilesRnn_7060926234834
// MI455X (gfx1250) — matrix operand order repaired and checked by reading, not compiled, not run on hardware
//
#include <hip/hip_runtime.h>

// ---------------------------------------------------------------------------
// Conditional SMILES RNN: 3-layer LSTM, H=1024, B=256, T=128, greedy decode.
// Persistent wave32 WMMA kernel for gfx1250 (MI455X).
//   - bf16 weights packed once per call (~51MB -> L2-resident, 192MB L2)
//   - v_wmma_f32_16x16x32_bf16 for all GEMMs, f32 accumulation
//   - weight k-slabs DMA'd into LDS via TDM (tensor_load_to_lds, TENSORcnt),
//     double-buffered; all 8 waves share them via ds_load_b128 (8x L2 cut)
//   - cell state c kept in VGPRs across all 128 timesteps
//   - grid-wide arrive/wait barrier (4 per step) for the sequential chain
// ---------------------------------------------------------------------------

#define BATCH  256
#define HID    1024
#define NPROP  4
#define NLAYER 3
#define SEQT   128
#define NOUT   47
#define NVOCAB 47

#define K0PAD  1152                 // H+P=1028 padded so KL0 % 128 == 0
#define KL0    (K0PAD + HID)        // 2176  (layer0: [x0 | h0])
#define KLR    (HID + HID)          // 2048  (layers 1,2: [h_below | h_self])
#define XP     (KL0 + KLR + KLR)    // 6272  per-batch-row pitch of Xbuf
#define NWG    64
#define TPB    256                  // 8 waves of 32

#define SLABK   128                 // k-elements per TDM slab (4 k-tiles)
#define SLABROW 272                 // LDS row pitch: 256B data + 16B pad
#define SLABSZ  (64 * SLABROW)      // 17408 B per slab buffer

typedef __attribute__((ext_vector_type(16))) __bf16 v16bf;
typedef __attribute__((ext_vector_type(8)))  __bf16 v8bf;
typedef __attribute__((ext_vector_type(8)))  float  v8f;
typedef unsigned int u32x4 __attribute__((ext_vector_type(4)));
typedef int          i32x8 __attribute__((ext_vector_type(8)));
typedef int          i32x4 __attribute__((ext_vector_type(4)));

#if defined(__gfx1250__) && __has_builtin(__builtin_amdgcn_tensor_load_to_lds) && \
    __has_builtin(__builtin_amdgcn_s_wait_tensorcnt)
#define USE_TDM 1
#endif

__device__ __forceinline__ float fsig(float x) { return 1.0f / (1.0f + __expf(-x)); }
__device__ __forceinline__ float ftanh_fast(float x) { return 2.0f * fsig(2.0f * x) - 1.0f; }

// Load a 16-element bf16 fragment as two contiguous 16B / 8-elem chunks.
// A-fragment (16x32): per-lane chunks at k = 16*half + {0..7, 8..15}  (stride 8)
// B-fragment (32x16): per-lane chunks at k = 16*half + {0..7, 8..15}  (stride 8)
__device__ __forceinline__ v16bf load_frag16(const __bf16* __restrict__ p, int stride2) {
  v8bf lo = *(const v8bf*)p;
  v8bf hi = *(const v8bf*)(p + stride2);
  v16bf r;
#pragma unroll
  for (int i = 0; i < 8; ++i) { r[i] = lo[i]; r[8 + i] = hi[i]; }
  return r;
}

__device__ __forceinline__ v8f wmma_bf16(v16bf a, v16bf b, v8f c) {
  return __builtin_amdgcn_wmma_f32_16x16x32_bf16(false, a, false, b, (short)0, c,
                                                 false, false);
}

#ifdef USE_TDM
// Issue a TDM load of a 64-row x SLABK-col bf16 tile (row stride = K elements)
// from global into LDS at byte offset ldsByte.  D# per CDNA5 ISA 8.3/8.4.
// pad_enable inserts 16B after every 256B row -> LDS pitch 272B (bank rotate).
// This toolchain's builtin takes 6 args: (g0 v4u, g1 v8i, g2 v4i, g3 v4i,
// extra v8i, cpol i32).
__device__ __forceinline__ void tdm_issue_slab(const __bf16* gaddr, unsigned ldsByte,
                                               int K) {
  unsigned long long ga = (unsigned long long)(size_t)gaddr;
  u32x4 g0;
  g0[0] = 1u;                                          // count=1, user mode
  g0[1] = ldsByte;                                     // lds_addr (bytes)
  g0[2] = (unsigned)(ga & 0xFFFFFFFFu);                // global_addr[31:0]
  g0[3] = (unsigned)((ga >> 32) & 0x01FFFFFFu) | (2u << 30);  // addr[56:32]|type=2
  i32x8 g1;
  g1[0] = (int)0x07510000u;   // data_size=2B, pad_en=1, pad_interval=256B, pad=16B
  g1[1] = (int)(((unsigned)K & 0xFFFFu) << 16);        // tensor_dim0[15:0]
  g1[2] = (int)((((unsigned)K >> 16) & 0xFFFFu) | (64u << 16)); // dim0 hi | dim1=64
  g1[3] = (int)((unsigned)SLABK << 16);                // tile_dim0 = 128 elements
  g1[4] = 64;                                          // tile_dim1=64 rows, dim2=0
  g1[5] = K;                                           // tensor_dim0_stride (elems)
  g1[6] = 0;
  g1[7] = 0;
  i32x4 z4;
  z4[0] = 0; z4[1] = 0; z4[2] = 0; z4[3] = 0;
  i32x8 z8;
  z8[0] = 0; z8[1] = 0; z8[2] = 0; z8[3] = 0;
  z8[4] = 0; z8[5] = 0; z8[6] = 0; z8[7] = 0;
  __builtin_amdgcn_tensor_load_to_lds(g0, g1, z4, z4, z8, 0);
}
#endif

__device__ __forceinline__ void grid_barrier(int* __restrict__ syncp) {
  __syncthreads();
  __threadfence();
  if (threadIdx.x == 0) {
    int* cnt = syncp;
    int* gen = syncp + 32;
    int g = __hip_atomic_load(gen, __ATOMIC_ACQUIRE, __HIP_MEMORY_SCOPE_AGENT);
    int v = __hip_atomic_fetch_add(cnt, 1, __ATOMIC_ACQ_REL, __HIP_MEMORY_SCOPE_AGENT);
    if (v == NWG - 1) {
      __hip_atomic_store(cnt, 0, __ATOMIC_RELAXED, __HIP_MEMORY_SCOPE_AGENT);
      __hip_atomic_fetch_add(gen, 1, __ATOMIC_RELEASE, __HIP_MEMORY_SCOPE_AGENT);
    } else {
      while (__hip_atomic_load(gen, __ATOMIC_ACQUIRE, __HIP_MEMORY_SCOPE_AGENT) == g) {
        __builtin_amdgcn_s_sleep(2);
      }
    }
  }
  __syncthreads();
  __threadfence();
}

// ---------------------------------------------------------------------------
// Prologue 1: small tables + Xbuf init (everything bf16 / packed order)
// ---------------------------------------------------------------------------
__global__ void k_prep(const float* __restrict__ props, const float* __restrict__ emb,
                       const float* __restrict__ W_dec, const float* __restrict__ b_dec,
                       const float* __restrict__ b_ih, const float* __restrict__ b_hh,
                       int* __restrict__ syncp, float* __restrict__ biasP,
                       float* __restrict__ bdecP, __bf16* __restrict__ embB,
                       __bf16* __restrict__ WdecP, __bf16* __restrict__ Xbuf) {
  long idx = (long)blockIdx.x * 256 + threadIdx.x;
  if (idx < 64) { syncp[idx] = 0; return; }
  idx -= 64;
  if (idx < 3 * 4096) {                       // bias, packed (wg,gate,j) order
    int l = (int)(idx / 4096), rr = (int)(idx % 4096);
    int wg = rr >> 6, q = rr & 63;
    int n = (q >> 4) * HID + wg * 16 + (q & 15);
    biasP[idx] = b_ih[l * 4096 + n] + b_hh[l * 4096 + n];
    return;
  }
  idx -= 3 * 4096;
  if (idx < 48) { bdecP[idx] = (idx < NOUT) ? b_dec[idx] : -1.0e30f; return; }
  idx -= 48;
  if (idx < (long)NVOCAB * HID) { embB[idx] = (__bf16)emb[idx]; return; }
  idx -= (long)NVOCAB * HID;
  if (idx < 48L * HID) {                      // W_dec padded to 48 rows, bf16
    int row = (int)(idx >> 10), col = (int)(idx & 1023);
    WdecP[idx] = (row < NOUT) ? (__bf16)W_dec[row * HID + col] : (__bf16)0.0f;
    return;
  }
  idx -= 48L * HID;
  if (idx < 2L * BATCH * XP) {                // zero Xbuf; parity0 x0 = [emb[0]|props]
    long pp = idx / ((long)BATCH * XP);
    long rem = idx - pp * (long)BATCH * XP;
    int b = (int)(rem / XP), col = (int)(rem % XP);
    float v = 0.0f;
    if (pp == 0 && col < HID + NPROP)
      v = (col < HID) ? emb[col] : props[b * NPROP + (col - HID)];
    Xbuf[idx] = (__bf16)v;
  }
}

// ---------------------------------------------------------------------------
// Prologue 2: pack [W_ih | W_hh] -> bf16, rows reordered so WG `wg` owns rows
// [wg*64, wg*64+64) = gates {i,f,g,o} x 16 hidden units (j = wg*16 + jj).
// ---------------------------------------------------------------------------
__global__ void k_pack_weights(const float* __restrict__ W_ih0,
                               const float* __restrict__ W_ihr,
                               const float* __restrict__ W_hh,
                               __bf16* __restrict__ WpAll) {
  size_t idx = (size_t)blockIdx.x * 256 + threadIdx.x;
  const size_t S0 = (size_t)4096 * KL0;
  const size_t S1 = (size_t)4096 * KLR;
  if (idx >= S0 + 2 * S1) return;
  int l; size_t local;
  if (idx < S0)           { l = 0; local = idx; }
  else if (idx < S0 + S1) { l = 1; local = idx - S0; }
  else                    { l = 2; local = idx - S0 - S1; }
  const int K   = (l == 0) ? KL0 : KLR;
  const int kin = (l == 0) ? K0PAD : HID;
  int r = (int)(local / K);
  int c = (int)(local % K);
  int wg = r >> 6, q = r & 63;
  int n = (q >> 4) * HID + wg * 16 + (q & 15);   // original gate-row
  float v;
  if (c < kin) {
    if (l == 0) v = (c < HID + NPROP) ? W_ih0[(size_t)n * (HID + NPROP) + c] : 0.0f;
    else        v = W_ihr[((size_t)(l - 1) * 4096 + n) * HID + c];
  } else {
    v = W_hh[((size_t)l * 4096 + n) * HID + (c - kin)];
  }
  WpAll[idx] = (__bf16)v;
}

// ---------------------------------------------------------------------------
// Persistent LSTM kernel: 64 WGs x 8 waves. Wave w owns batch rows
// [32w, 32w+32) (2 M-tiles); WG owns hidden slice j in [wg*16, wg*16+16).
// ---------------------------------------------------------------------------
__global__ __launch_bounds__(TPB, 1) void lstm_persistent(
    const float* __restrict__ props, const __bf16* __restrict__ embB,
    const float* __restrict__ biasP, const float* __restrict__ bdecP,
    const __bf16* __restrict__ WdecP, const __bf16* __restrict__ WpAll,
    __bf16* __restrict__ Xbuf, __bf16* __restrict__ Htop,
    int* __restrict__ syncp, float* __restrict__ outLogits,
    float* __restrict__ outH, float* __restrict__ outC) {
#ifdef USE_TDM
  __shared__ __align__(16) unsigned char sB[2][SLABSZ];  // double-buffered W slab
#endif
  __shared__ float sLog[16 * 48];
  __shared__ int   sTok[16];

  const int tid  = threadIdx.x;
  const int wg   = blockIdx.x;
  const int wave = tid >> 5;
  const int lane = tid & 31;
  const int lh   = lane >> 4;   // half of wave (K/M-split per ISA layouts)
  const int ln   = lane & 15;   // N / row-in-tile index

  const __bf16* Wp0 = WpAll;
  const __bf16* Wp1 = WpAll + (size_t)4096 * KL0;
  const __bf16* Wp2 = Wp1 + (size_t)4096 * KLR;
  const __bf16* Wp[NLAYER]   = {Wp0, Wp1, Wp2};
  const int     Kl[NLAYER]   = {KL0, KLR, KLR};
  const int     xcol[NLAYER] = {0, KL0, KL0 + KLR};
  const int     hoff[NLAYER] = {K0PAD, HID, HID};

  // Per-lane gate biases (depend only on j = wg*16 + ln)
  float biasr[NLAYER][4];
#pragma unroll
  for (int l = 0; l < NLAYER; ++l)
#pragma unroll
    for (int g = 0; g < 4; ++g)
      biasr[l][g] = biasP[l * 4096 + wg * 64 + g * 16 + ln];

  // Persistent cell state, in registers for the whole sequence
  v8f cst[NLAYER][2];
#pragma unroll
  for (int l = 0; l < NLAYER; ++l)
#pragma unroll
    for (int mt = 0; mt < 2; ++mt)
#pragma unroll
      for (int r = 0; r < 8; ++r) cst[l][mt][r] = 0.0f;

  for (int t = 0; t < SEQT; ++t) {
    const int p = t & 1;
    const __bf16* Xr = Xbuf + (size_t)p * BATCH * XP;        // read parity
    __bf16* Xw = Xbuf + (size_t)(1 - p) * BATCH * XP;        // next-step parity
    __bf16* Xs = (__bf16*)Xr;                                // same-step writes

#pragma unroll
    for (int l = 0; l < NLAYER; ++l) {
      const __bf16* W = Wp[l];
      const int K  = Kl[l];
      const int nslab = (K / SLABK);

      v8f acc[2][4];
#pragma unroll
      for (int mt = 0; mt < 2; ++mt)
#pragma unroll
        for (int nt = 0; nt < 4; ++nt)
#pragma unroll
          for (int r = 0; r < 8; ++r) acc[mt][nt][r] = 0.0f;

      const __bf16* arow0 = Xr + (size_t)(wave * 32 + ln) * XP + xcol[l];
      const __bf16* arow1 = arow0 + (size_t)16 * XP;
      const __bf16* brow  = W + (size_t)(wg * 64 + ln) * K;
      const __bf16* wbase = W + (size_t)(wg * 64) * K;       // WG's 64-row slab
      (void)brow; (void)wbase;

#ifdef USE_TDM
      if (wave == 0) {                       // pre-issue two slabs (TENSORcnt)
        tdm_issue_slab(wbase, (unsigned)(size_t)&sB[0][0], K);
        if (nslab > 1)
          tdm_issue_slab(wbase + SLABK, (unsigned)(size_t)&sB[1][0], K);
      }
#endif
      for (int s = 0; s < nslab; ++s) {
        const int q = s & 1;
#ifdef USE_TDM
        if (wave == 0) {
          if (s < nslab - 1) __builtin_amdgcn_s_wait_tensorcnt(1);
          else               __builtin_amdgcn_s_wait_tensorcnt(0);
        }
        __syncthreads();                     // slab s visible to all waves
#endif
#pragma unroll
        for (int kk = 0; kk < SLABK / 32; ++kk) {
          const int kb = (s * (SLABK / 32) + kk) * 32;
          v16bf a0 = load_frag16(arow0 + kb + lh * 16, 8);
          v16bf a1 = load_frag16(arow1 + kb + lh * 16, 8);
          __builtin_prefetch(arow0 + kb + SLABK, 0, 1);      // global_prefetch_b8
#ifdef USE_TDM
          const unsigned char* bb = &sB[q][0] + (size_t)ln * SLABROW + kk * 64 + lh * 32;
#pragma unroll
          for (int nt = 0; nt < 4; ++nt) {
            v16bf b = load_frag16((const __bf16*)(bb + (size_t)nt * 16 * SLABROW), 8);
            acc[0][nt] = wmma_bf16(a0, b, acc[0][nt]);
            acc[1][nt] = wmma_bf16(a1, b, acc[1][nt]);
          }
#else
#pragma unroll
          for (int nt = 0; nt < 4; ++nt) {
            v16bf b = load_frag16(brow + (size_t)nt * 16 * K + kb + lh * 16, 8);
            acc[0][nt] = wmma_bf16(a0, b, acc[0][nt]);
            acc[1][nt] = wmma_bf16(a1, b, acc[1][nt]);
          }
#endif
        }
#ifdef USE_TDM
        __syncthreads();                     // all waves done reading buf q
        if (wave == 0 && s + 2 < nslab)
          tdm_issue_slab(wbase + (size_t)(s + 2) * SLABK,
                         (unsigned)(size_t)&sB[q][0], K);
#endif
      }

      // LSTM elementwise entirely in accumulator registers
#pragma unroll
      for (int mt = 0; mt < 2; ++mt) {
        const int mbase = wave * 32 + mt * 16 + lh * 8;
        const int j = wg * 16 + ln;
#pragma unroll
        for (int r = 0; r < 8; ++r) {
          float gi = acc[mt][0][r] + biasr[l][0];
          float gf = acc[mt][1][r] + biasr[l][1];
          float gg = acc[mt][2][r] + biasr[l][2];
          float go = acc[mt][3][r] + biasr[l][3];
          float cn = fsig(gf) * cst[l][mt][r] + fsig(gi) * ftanh_fast(gg);
          float hn = fsig(go) * ftanh_fast(cn);
          cst[l][mt][r] = cn;
          const int m = mbase + r;
          __bf16 hb = (__bf16)hn;
          Xw[(size_t)m * XP + xcol[l] + hoff[l] + j] = hb;   // own h, next step
          if (l < 2) Xs[(size_t)m * XP + xcol[l + 1] + j] = hb; // feed layer above
          else       Htop[(size_t)m * HID + j] = hb;            // feed decoder
          if (t == SEQT - 1) {
            outH[((size_t)l * BATCH + m) * HID + j] = hn;
            outC[((size_t)l * BATCH + m) * HID + j] = cn;
          }
        }
      }
      grid_barrier(syncp);
    }

    // ---- decode: logits = Htop @ Wdec^T + bdec, argmax, gather next x0 ----
    if (wg < 16) {
      if (wave < 3) {                          // 3 N-tiles cover 48 (padded 47)
        v8f dacc;
#pragma unroll
        for (int r = 0; r < 8; ++r) dacc[r] = 0.0f;
        const __bf16* ah = Htop + (size_t)(wg * 16 + ln) * HID;
        const __bf16* bw = WdecP + (size_t)(wave * 16 + ln) * HID;
        for (int kt = 0; kt < HID / 32; ++kt) {
          v16bf a = load_frag16(ah + kt * 32 + lh * 16, 8);
          v16bf b = load_frag16(bw + kt * 32 + lh * 16, 8);
          dacc = wmma_bf16(a, b, dacc);
        }
        const int n = wave * 16 + ln;
        const float bn = bdecP[n];
#pragma unroll
        for (int r = 0; r < 8; ++r) {
          float v = dacc[r] + bn;
          int mr = lh * 8 + r;                 // row within M-tile
          sLog[mr * 48 + n] = v;
          if (n < NOUT)
            outLogits[((size_t)(wg * 16 + mr) * SEQT + t) * NOUT + n] = v;
        }
      }
      __syncthreads();
      if (tid < 16) {                          // greedy argmax (first max wins)
        const float* row = sLog + tid * 48;
        float bv = row[0]; int bi = 0;
#pragma unroll
        for (int n = 1; n < 48; ++n) { float v = row[n]; if (v > bv) { bv = v; bi = n; } }
        sTok[tid] = bi;
      }
      __syncthreads();
#pragma unroll 1
      for (int rr = 0; rr < 16; ++rr) {        // x0[next] = emb[token] (bf16)
        const int mrow = wg * 16 + rr;
        const uint2* src = (const uint2*)(embB + (size_t)sTok[rr] * HID);
        uint2* dst = (uint2*)(Xw + (size_t)mrow * XP);
        dst[tid] = src[tid];                   // 256 thr x 8B = 1024 bf16
      }
      if (tid < 64) {                          // x0[1024..1027] = props (bf16)
        const int rr = tid >> 2, pi = tid & 3;
        const int mrow = wg * 16 + rr;
        Xw[(size_t)mrow * XP + HID + pi] = (__bf16)props[mrow * NPROP + pi];
      }
    }
    grid_barrier(syncp);
  }
}

// ---------------------------------------------------------------------------
extern "C" void kernel_launch(void* const* d_in, const int* in_sizes, int n_in,
                              void* d_out, int out_size, void* d_ws, size_t ws_size,
                              hipStream_t stream) {
  (void)in_sizes; (void)n_in; (void)out_size;
  const float* props = (const float*)d_in[1];
  const float* emb   = (const float*)d_in[2];
  const float* W_dec = (const float*)d_in[3];
  const float* b_dec = (const float*)d_in[4];
  const float* W_ih0 = (const float*)d_in[5];
  const float* W_ihr = (const float*)d_in[6];
  const float* W_hh  = (const float*)d_in[7];
  const float* b_ih  = (const float*)d_in[8];
  const float* b_hh  = (const float*)d_in[9];

  char* ws = (char*)d_ws;
  size_t cur = 0;
  auto alloc = [&](size_t bytes) {
    size_t o = cur; cur += (bytes + 255) & ~(size_t)255; return o;
  };
  int*    syncp = (int*)   (ws + alloc(64 * sizeof(int)));
  float*  biasP = (float*) (ws + alloc((size_t)3 * 4096 * sizeof(float)));
  float*  bdecP = (float*) (ws + alloc(48 * sizeof(float)));
  __bf16* embB  = (__bf16*)(ws + alloc((size_t)NVOCAB * HID * 2));
  __bf16* WdecP = (__bf16*)(ws + alloc((size_t)48 * HID * 2));
  __bf16* WpAll = (__bf16*)(ws + alloc(((size_t)4096 * KL0 + 2 * (size_t)4096 * KLR) * 2));
  __bf16* Xbuf  = (__bf16*)(ws + alloc((size_t)2 * BATCH * XP * 2));
  __bf16* Htop  = (__bf16*)(ws + alloc((size_t)BATCH * HID * 2));
  if (ws_size < cur) return;   // ~59 MB of scratch required

  float* outLogits = (float*)d_out;
  float* outH = outLogits + (size_t)BATCH * SEQT * NOUT;
  float* outC = outH + (size_t)NLAYER * BATCH * HID;

  long prepN = 64 + 3L * 4096 + 48 + (long)NVOCAB * HID + 48L * HID + 2L * BATCH * XP;
  k_prep<<<(int)((prepN + 255) / 256), 256, 0, stream>>>(
      props, emb, W_dec, b_dec, b_ih, b_hh, syncp, biasP, bdecP, embB, WdecP, Xbuf);

  size_t packN = (size_t)4096 * KL0 + 2 * (size_t)4096 * KLR;
  k_pack_weights<<<(int)((packN + 255) / 256), 256, 0, stream>>>(W_ih0, W_ihr, W_hh, WpAll);

  lstm_persistent<<<NWG, TPB, 0, stream>>>(props, embB, biasP, bdecP, WdecP, WpAll,
                                           Xbuf, Htop, syncp, outLogits, outH, outC);
}
